// DySepConvAtten_37555194036865
// MI455X (gfx1250) — compile-verified, matrix operand order checked by reading, not run on hardware
//
#include <hip/hip_runtime.h>

// Problem constants (B=2048, N=64, C=256, K=3)
#define BATCH_N 64
#define CH      256
#define KW      3
#define OUTW    67      // N + K
#define OUTW_PAD 80

typedef __attribute__((ext_vector_type(8)))  __bf16 bf16x8;
typedef __attribute__((ext_vector_type(16))) __bf16 v16bf;
typedef __attribute__((ext_vector_type(8)))  float  v8f;

// Dynamic-LDS layout (bytes). Total 184,128 B of the 320 KB WGP pool.
// NOTE: kernel has no static __shared__, so dynamic LDS starts at wave LDS
// offset 0 — the raw byte offsets below are valid LDS addresses for the
// async-copy instructions.
#define OFF_Q    0          // 64*256 bf16 = 32768   (A of GEMM1)
#define OFF_W    32768      // 80*256 bf16 = 40960   (B of GEMM1, row=n, col=k)
#define OFF_VP   73728      // 64*256 f32  = 65536   (value, later reused for point)
#define OFF_PW   139264     // 64*64  bf16 = 8192    (point_w, A of GEMM2)
#define OFF_DW   147456     // 64*4   f32  = 1024    (depth_w)
#define OFF_DT   148480     // 256*64 bf16 = 32768   (depth transposed, B of GEMM2)
#define OFF_RED  181248     // 64*4*2 f32  = 2048    (LN partials)
#define OFF_ST   183296     // 64*2   f32  = 512     (LN mu / rsigma)
#define OFF_BIAS 183808     // 80     f32  = 320     (padded b_lin)
#define SMEM_BYTES 184128

// Build a 16x32 bf16 WMMA operand fragment from a row-major row pointer.
// Per ISA layout: lane half h=lane>>4 owns K = {k0+8h .. k0+8h+7} U {k0+16+8h .. k0+23+8h}.
__device__ __forceinline__ v16bf load_frag(const __bf16* __restrict__ row, int k0, int h) {
  union { v16bf v; bf16x8 half2[2]; } u;
  u.half2[0] = *(const bf16x8*)(row + k0 + 8 * h);
  u.half2[1] = *(const bf16x8*)(row + k0 + 16 + 8 * h);
  return u.v;
}

__global__ __launch_bounds__(256)
void dysepconv_fused_kernel(const float* __restrict__ query,
                            const float* __restrict__ value,
                            const float* __restrict__ W_lin,
                            const float* __restrict__ b_lin,
                            const float* __restrict__ gamma,
                            const float* __restrict__ beta,
                            float* __restrict__ out) {
  extern __shared__ __align__(16) unsigned char smem[];
  __bf16* sQ   = (__bf16*)(smem + OFF_Q);
  __bf16* sW   = (__bf16*)(smem + OFF_W);
  float*  sVP  = (float*) (smem + OFF_VP);   // value, then point (reuse)
  __bf16* sPw  = (__bf16*)(smem + OFF_PW);
  float*  sDw  = (float*) (smem + OFF_DW);
  __bf16* sDT  = (__bf16*)(smem + OFF_DT);
  float*  sRed = (float*) (smem + OFF_RED);
  float*  sSt  = (float*) (smem + OFF_ST);
  float*  sBias= (float*) (smem + OFF_BIAS);

  const int tid  = threadIdx.x;
  const int wave = tid >> 5;
  const int lane = tid & 31;
  const int lm   = lane & 15;   // row (A) / col (B,C) within 16x16 tile
  const int h    = lane >> 4;
  const int b    = blockIdx.x;

  const float* Qb = query + (size_t)b * (BATCH_N * CH);
  const float* Vb = value + (size_t)b * (BATCH_N * CH);
  float*       Ob = out   + (size_t)b * (BATCH_N * CH);

  // ---- Stage 0a: kick off async HBM->LDS copy of value (consumed in stage 2).
  // 64 KB streams in on ASYNCcnt while GEMM1's WMMAs execute.
  {
    #pragma unroll
    for (int it = 0; it < (BATCH_N * CH * 4) / (256 * 16); ++it) {  // 16 iters
      const int idx = it * 256 + tid;                 // 16-byte chunk id
      const unsigned lds_addr = OFF_VP + idx * 16;    // LDS byte address
      const unsigned goff     = idx * 16;             // global byte offset
      asm volatile("global_load_async_to_lds_b128 %0, %1, %2"
                   :: "v"(lds_addr), "v"(goff), "s"(Vb) : "memory");
    }
  }

  // ---- Stage 0b: synchronous loads: Q (f32->bf16), W (f32->bf16 + pad), bias ----
  for (int i = tid; i < BATCH_N * CH / 4; i += 256) {
    float4 q4 = ((const float4*)Qb)[i];
    sQ[4*i+0] = (__bf16)q4.x; sQ[4*i+1] = (__bf16)q4.y;
    sQ[4*i+2] = (__bf16)q4.z; sQ[4*i+3] = (__bf16)q4.w;
  }
  for (int i = tid; i < OUTW * CH / 4; i += 256) {
    float4 w4 = ((const float4*)W_lin)[i];
    sW[4*i+0] = (__bf16)w4.x; sW[4*i+1] = (__bf16)w4.y;
    sW[4*i+2] = (__bf16)w4.z; sW[4*i+3] = (__bf16)w4.w;
  }
  for (int i = tid; i < (OUTW_PAD - OUTW) * CH; i += 256)
    sW[OUTW * CH + i] = (__bf16)0.f;
  if (tid < OUTW_PAD)
    sBias[tid] = (tid < OUTW) ? b_lin[tid] : 0.f;
  __syncthreads();

  // ---- Stage 1 (GEMM1): dyw = Q(64x256) x W^T(256x80); 4x5 tiles over 8 waves ----
  for (int t = wave; t < 4 * 5; t += 8) {
    const int tM = t / 5, tN = t % 5;
    const __bf16* arow = sQ + (size_t)(tM * 16 + lm) * CH;  // A[m][k]
    const __bf16* brow = sW + (size_t)(tN * 16 + lm) * CH;  // B[k][n] = W[n][k]
    v8f acc = {};
    #pragma unroll
    for (int k0 = 0; k0 < CH; k0 += 32) {
      v16bf a  = load_frag(arow, k0, h);
      v16bf bb = load_frag(brow, k0, h);
      acc = __builtin_amdgcn_wmma_f32_16x16x32_bf16(false, a, false, bb,
                                                    (short)0, acc, false, false);
    }
    const int gn = tN * 16 + lm;
    const float bias = sBias[gn];
    #pragma unroll
    for (int v = 0; v < 8; ++v) {
      const int gm = tM * 16 + v + 8 * h;     // C layout: M = v (+8 for lanes 16..31)
      const float val = acc[v] + bias;
      if (gn < KW)           sDw[gm * 4 + gn]          = val;          // depth_w
      else if (gn < OUTW)    sPw[gm * 64 + (gn - KW)]  = (__bf16)val;  // point_w
    }
  }
  // Drain the async value copy issued in stage 0a, then make it WG-visible.
  asm volatile("s_wait_asynccnt 0x0" ::: "memory");
  __syncthreads();

  // ---- Stage 2: depthwise 3-tap conv along C + ReLU, stored transposed bf16 ----
  for (int i = tid; i < BATCH_N * CH; i += 256) {
    const int n = i >> 8, c = i & 255;
    const float* vr = sVP + n * CH;
    const float x0 = (c > 0)      ? vr[c - 1] : 0.f;
    const float x1 = vr[c];
    const float x2 = (c < CH - 1) ? vr[c + 1] : 0.f;
    float d = sDw[n * 4 + 0] * x0 + sDw[n * 4 + 1] * x1 + sDw[n * 4 + 2] * x2;
    sDT[c * 64 + n] = (__bf16)fmaxf(d, 0.f);   // depth^T[c][n]
  }
  __syncthreads();

  // ---- Stage 3 (GEMM2): point = point_w(64x64) x depth(64x256) -> sVP (reuse) ----
  for (int t = wave; t < 4 * 16; t += 8) {
    const int tM = t >> 4, tN = t & 15;
    const __bf16* arow = sPw + (size_t)(tM * 16 + lm) * 64;  // A[m][k=n]
    const __bf16* brow = sDT + (size_t)(tN * 16 + lm) * 64;  // B[k][c] = depth^T[c][k]
    v8f acc = {};
    #pragma unroll
    for (int k0 = 0; k0 < 64; k0 += 32) {
      v16bf a  = load_frag(arow, k0, h);
      v16bf bb = load_frag(brow, k0, h);
      acc = __builtin_amdgcn_wmma_f32_16x16x32_bf16(false, a, false, bb,
                                                    (short)0, acc, false, false);
    }
    #pragma unroll
    for (int v = 0; v < 8; ++v)
      sVP[(tM * 16 + v + 8 * h) * CH + tN * 16 + lm] = acc[v];
  }
  __syncthreads();

  // ---- Stage 4: LayerNorm over C (two-level LDS reduction) ----
  {
    const int row = tid >> 2, part = tid & 3;
    const float* pr = sVP + row * CH + part * 64;
    float s = 0.f, ss = 0.f;
    #pragma unroll 8
    for (int j = 0; j < 64; ++j) { const float x = pr[j]; s += x; ss += x * x; }
    sRed[(row * 4 + part) * 2 + 0] = s;
    sRed[(row * 4 + part) * 2 + 1] = ss;
  }
  __syncthreads();
  if (tid < 64) {
    float s = 0.f, ss = 0.f;
    #pragma unroll
    for (int p = 0; p < 4; ++p) {
      s  += sRed[(tid * 4 + p) * 2 + 0];
      ss += sRed[(tid * 4 + p) * 2 + 1];
    }
    const float mu  = s * (1.f / CH);
    const float var = ss * (1.f / CH) - mu * mu;
    sSt[tid * 2 + 0] = mu;
    sSt[tid * 2 + 1] = rsqrtf(var + 1e-5f);
  }
  __syncthreads();
  {
    const float g  = gamma[tid];
    const float be = beta[tid];
    for (int n = 0; n < BATCH_N; ++n) {
      const float x = sVP[n * CH + tid];
      Ob[n * CH + tid] = (x - sSt[2 * n]) * sSt[2 * n + 1] * g + be;  // coalesced
    }
  }
}

extern "C" void kernel_launch(void* const* d_in, const int* in_sizes, int n_in,
                              void* d_out, int out_size, void* d_ws, size_t ws_size,
                              hipStream_t stream) {
  const float* query = (const float*)d_in[0];
  const float* value = (const float*)d_in[1];
  const float* W_lin = (const float*)d_in[2];
  const float* b_lin = (const float*)d_in[3];
  const float* gamma = (const float*)d_in[4];
  const float* beta  = (const float*)d_in[5];
  float* out = (float*)d_out;
  const int B = in_sizes[0] / (BATCH_N * CH);
  dysepconv_fused_kernel<<<dim3(B), dim3(256), SMEM_BYTES, stream>>>(
      query, value, W_lin, b_lin, gamma, beta, out);
}
